// HeadWisePhybridAttention_8907762172226
// MI455X (gfx1250) — matrix operand order repaired and checked by reading, not compiled, not run on hardware
//
#include <hip/hip_runtime.h>
#include <cstdint>
#include <cstddef>

// ---------------- problem dims ----------------
#define B_    2
#define S_    2048
#define HID_  2048
#define H_    16
#define KVH_  8
#define D_    128
#define ATTNH_ 8
#define TNH_   8
#define FIRST_ 64
#define LAST_  512
#define SCALE_ 0.08838834764831845f   // 1/sqrt(128)

// ---------------- CDNA5 async global->LDS (guarded) ----------------
#if defined(__has_builtin)
#if __has_builtin(__builtin_amdgcn_global_load_async_to_lds_b128) && \
    __has_builtin(__builtin_amdgcn_s_wait_asynccnt)
#define USE_ASYNC_LDS 1
#endif
#endif
#ifndef USE_ASYNC_LDS
#define USE_ASYNC_LDS 0
#endif

// ---------------- types ----------------
typedef __attribute__((ext_vector_type(16))) __bf16        bfrag;
typedef __attribute__((ext_vector_type(8)))  float         v8f;
typedef __attribute__((ext_vector_type(4)))  unsigned int  v4u;
typedef __attribute__((ext_vector_type(4)))  int           v4i;

union FragU { bfrag v; v4u q[2]; };

__device__ __forceinline__ unsigned short f2bf(float f) {
  unsigned u = __float_as_uint(f);
  unsigned r = u + 0x7FFFu + ((u >> 16) & 1u);   // round-to-nearest-even
  return (unsigned short)(r >> 16);
}
__device__ __forceinline__ float bf2f(unsigned short h) {
  return __uint_as_float(((unsigned)h) << 16);
}
__device__ __forceinline__ v8f zero8() {
  v8f z;
#pragma unroll
  for (int i = 0; i < 8; ++i) z[i] = 0.0f;
  return z;
}

// 16-byte global -> LDS mover: async on CDNA5, sync fallback otherwise.
__device__ __forceinline__ void copy16_g2l(const unsigned short* g, unsigned short* l) {
#if USE_ASYNC_LDS
  __builtin_amdgcn_global_load_async_to_lds_b128(
      (__attribute__((address_space(1))) v4i*)g,
      (__attribute__((address_space(3))) v4i*)l, 0, 0);
#else
  *(v4u*)l = *(const v4u*)g;
#endif
}
__device__ __forceinline__ void async_wait0() {
#if USE_ASYNC_LDS
  __builtin_amdgcn_s_wait_asynccnt(0);
#endif
}

// A-matrix 16x32 bf16 fragment (row-major source, row=lane&15, K contiguous).
// ISA layout: lanes 0-15 hold K {0..7,16..23}, lanes 16-31 hold K {8..15,24..31}.
__device__ __forceinline__ bfrag load_frag_a(const unsigned short* base, int ld, int lane) {
  const int row  = lane & 15;
  const int half = lane >> 4;
  FragU f;
  const unsigned short* p = base + (size_t)row * ld + 8 * half;
  f.q[0] = *(const v4u*)(p);        // K = 8*half .. 8*half+7
  f.q[1] = *(const v4u*)(p + 16);   // K = 16+8*half .. 16+8*half+7
  return f.v;
}

// B-matrix 32x16 bf16 fragment: column n = lane&15, source row-major over K
// (i.e. row n of a [N,K] matrix). lanes 0-15: K {0..7,16..23}, lanes 16-31: K {8..15,24..31}.
__device__ __forceinline__ bfrag load_frag_b(const unsigned short* base, int ld, int lane) {
  const int row  = lane & 15;
  const int half = lane >> 4;
  FragU f;
  const unsigned short* p = base + (size_t)row * ld + 8 * half;
  f.q[0] = *(const v4u*)(p);        // K = 8*half .. 8*half+7
  f.q[1] = *(const v4u*)(p + 16);   // K = 16+8*half .. 16+8*half+7
  return f.v;
}

__device__ __forceinline__ v8f wmma_bf16(bfrag a, bfrag b, v8f c) {
  return __builtin_amdgcn_wmma_f32_16x16x32_bf16(false, a, false, b, (short)0, c, false, false);
}

// ---------------- fp32 -> bf16 pack ----------------
__global__ void pack_bf16_kernel(const float* __restrict__ src,
                                 unsigned short* __restrict__ dst, int n) {
  int i = blockIdx.x * blockDim.x + threadIdx.x;
  if (i < n) dst[i] = f2bf(src[i]);
}

// ---------------- GEMM: out[M,N] = A[M,K] (bf16) @ W[N,K]^T (bf16) + bias ----------------
// block = 256 threads (8 waves), block tile 128(M) x 64(N), wave tile 16x64.
// B tile (64 x 64) double-buffered in LDS via async global->LDS copies.
#define BT_LD 72   // 64 + 8 pad halves (144B row stride, 16B aligned)

template <bool OUT_BF16>
__global__ __launch_bounds__(256)
void gemm_wmma_kernel(const unsigned short* __restrict__ A,
                      const unsigned short* __restrict__ W,
                      const float* __restrict__ bias,
                      unsigned short* __restrict__ outb,
                      float* __restrict__ outf,
                      int M, int N, int K) {
  __shared__ unsigned short Bt[2][64][BT_LD];   // 18,432 B

  const int lane = threadIdx.x & 31;
  const int wave = threadIdx.x >> 5;
  const int m0 = blockIdx.y * 128 + wave * 16;
  const int n0 = blockIdx.x * 64;

  v8f acc[4];
#pragma unroll
  for (int j = 0; j < 4; ++j) acc[j] = zero8();

  // issue the 64x64 B tile for k-chunk kc into LDS buffer `buf`
  auto issueB = [&](int kc, int buf) {
#pragma unroll
    for (int i = 0; i < 2; ++i) {
      const int idx = (int)threadIdx.x + i * 256;  // 0..511
      const int r = idx >> 3, seg = idx & 7;       // 64 rows x 8 x 16B
      copy16_g2l(W + (size_t)(n0 + r) * K + kc + seg * 8, &Bt[buf][r][seg * 8]);
    }
  };

  issueB(0, 0);

  const unsigned short* aBase = A + (size_t)m0 * K;
  for (int kc = 0; kc < K; kc += 64) {
    const int buf = (kc >> 6) & 1;
    async_wait0();        // our async writes to `buf` complete
    __syncthreads();      // all waves' writes to `buf` visible; prev reads of buf^1 done
    if (kc + 64 < K) issueB(kc + 64, buf ^ 1);

    __builtin_prefetch(aBase + kc + 1024, 0, 1);  // global_prefetch_b8
#pragma unroll
    for (int cc = 0; cc < 2; ++cc) {
      bfrag a = load_frag_a(aBase + kc + cc * 32, K, lane);
      bfrag bj[4];
#pragma unroll
      for (int j = 0; j < 4; ++j)
        bj[j] = load_frag_b(&Bt[buf][j * 16][cc * 32], BT_LD, lane);
#pragma unroll
      for (int j = 0; j < 4; ++j) acc[j] = wmma_bf16(a, bj[j], acc[j]);
    }
    __syncthreads();      // reads of `buf` done before it is overwritten again
  }

  const int half = lane >> 4, l15 = lane & 15;
#pragma unroll
  for (int j = 0; j < 4; ++j) {
    const int col = n0 + j * 16 + l15;
    const float bv = bias ? bias[col] : 0.0f;
#pragma unroll
    for (int v = 0; v < 8; ++v) {
      const int row = m0 + v + 8 * half;
      const float val = acc[j][v] + bv;
      if (OUT_BF16) outb[(size_t)row * N + col] = f2bf(val);
      else          outf[(size_t)row * N + col] = val;
    }
  }
}

// ---------------- RoPE + head permutation (bf16 in/out, fp32 math) ----------------
__global__ __launch_bounds__(128)
void rope_permute_kernel(const unsigned short* __restrict__ Qg,  // [B,S,H*D]
                         const unsigned short* __restrict__ Kg,  // [B,S,KVH*D]
                         const int* __restrict__ perm,
                         unsigned short* __restrict__ Qp,        // [B,S,H,D] (perm applied)
                         unsigned short* __restrict__ Kr) {      // [B,S,KVH,D]
  const int s = blockIdx.x;
  const int b = blockIdx.y;
  const int d = threadIdx.x;                 // 0..127
  const int fi = d & 63;
  const float LN_BASE = 13.815510557964274f; // ln(1e6)
  const float inv_freq = __expf(-((float)(2 * fi) / (float)D_) * LN_BASE);
  const float ang = (float)s * inv_freq;
  const float c = __cosf(ang), sn = __sinf(ang);
  const int d2 = (d + 64) & 127;
  const float sgn = (d < 64) ? -1.0f : 1.0f;  // rotate_half sign

  const size_t rowQ = ((size_t)b * S_ + s) * (size_t)(H_ * D_);
  const size_t rowK = ((size_t)b * S_ + s) * (size_t)(KVH_ * D_);

#pragma unroll
  for (int h = 0; h < H_; ++h) {
    const int src = perm[h];
    const float x  = bf2f(Qg[rowQ + src * D_ + d]);
    const float xr = bf2f(Qg[rowQ + src * D_ + d2]);
    Qp[rowQ + h * D_ + d] = f2bf(x * c + sgn * xr * sn);
  }
#pragma unroll
  for (int kv = 0; kv < KVH_; ++kv) {
    const float x  = bf2f(Kg[rowK + kv * D_ + d]);
    const float xr = bf2f(Kg[rowK + kv * D_ + d2]);
    Kr[rowK + kv * D_ + d] = f2bf(x * c + sgn * xr * sn);
  }
}

// ---------------- Attention (softmax flash / linear) ----------------
// block = 256 threads (8 waves); q-tile 128 (16 rows/wave); key tile 64.
#define KT_LD 136  // 128 + 8 pad halves (272B row stride, 16B aligned)
#define VT_LD 72
#define P_LD  72

__global__ __launch_bounds__(256)
void attn_wmma_kernel(const unsigned short* __restrict__ Qp,   // [B,S,H,D] bf16, rope+perm
                      const unsigned short* __restrict__ Kr,   // [B,S,KVH,D] bf16, rope
                      const unsigned short* __restrict__ Vv,   // [B,S,KVH,D] bf16
                      const unsigned char* __restrict__ policy,
                      const int* __restrict__ perm,
                      unsigned short* __restrict__ ctx,        // [B,S,H,D] bf16 (inv-perm slots)
                      float* __restrict__ out_tn,              // [B,S,TNH,D] f32
                      int linear_mode) {
  __shared__ unsigned short kt[64][KT_LD];      // K tile row-major: kt[ki][d]   (17,408 B)
  __shared__ unsigned short vt[D_][VT_LD];      // V tile transposed: vt[d][ki]  (18,432 B)
  __shared__ unsigned short pp[8][16][P_LD];    // per-wave P tiles (bf16)       (18,432 B)

  const int tid  = threadIdx.x;
  const int lane = tid & 31;
  const int wave = tid >> 5;
  const int half = lane >> 4, l15 = lane & 15;
  const int h    = (linear_mode ? ATTNH_ : 0) + blockIdx.y;
  const int b    = blockIdx.z;
  const int kvh  = perm[h] >> 1;                // GQA: repeat(kh,2)[perm[h]]
  const int qlo  = blockIdx.x * 128;
  const int q0   = qlo + wave * 16;
  const bool band = (!linear_mode) && (policy[h] == 0);

  // Q fragments held in registers for the whole kernel (16 rows x 128 d)
  bfrag qf[4];
  const unsigned short* qbase = Qp + (((size_t)b * S_ + q0) * H_ + h) * D_;
#pragma unroll
  for (int c = 0; c < 4; ++c) qf[c] = load_frag_a(qbase + c * 32, H_ * D_, lane);

  v8f acc[8];
#pragma unroll
  for (int dt = 0; dt < 8; ++dt) acc[dt] = zero8();
  float mreg[8], lreg[8];
#pragma unroll
  for (int v = 0; v < 8; ++v) { mreg[v] = -1e30f; lreg[v] = 0.0f; }

  const int kend = qlo + 127;
  for (int kb = 0; kb <= kend; kb += 64) {
    // block-uniform band skip: keep kb==0 (prefix) and the trailing band window
    if (band && kb > 0 && kb < qlo - 574) continue;

    __syncthreads();   // prior tile's LDS reads complete before overwrite

    // async-stage the 64x128 K tile into LDS (row-major)
#pragma unroll
    for (int i = 0; i < 4; ++i) {
      const int idx = tid + i * 256;           // 0..1023
      const int r = idx >> 4, seg = idx & 15;  // 64 rows x 16 x 16B
      copy16_g2l(Kr + (((size_t)b * S_ + (kb + r)) * KVH_ + kvh) * D_ + seg * 8,
                 &kt[r][seg * 8]);
    }

    // cooperative transpose-load of the 64x128 V tile into LDS
#pragma unroll
    for (int i = 0; i < 16; ++i) {
      const int idx = tid + i * 256;           // 0..4095 (pairs)
      const int ki = idx >> 6;
      const int dp = (idx & 63) * 2;
      const unsigned short* vp =
          Vv + (((size_t)b * S_ + (kb + ki)) * KVH_ + kvh) * D_ + dp;
      const unsigned int w = *(const unsigned int*)vp;
      vt[dp][ki]     = (unsigned short)(w & 0xFFFFu);
      vt[dp + 1][ki] = (unsigned short)(w >> 16);
    }

    async_wait0();     // K tile async writes complete
    __syncthreads();   // all waves' K/V tile data visible

    // scores: S = Q x K^T  (4 n-tiles of 16x16, depth 128) -- K from LDS
    v8f sc[4];
#pragma unroll
    for (int j = 0; j < 4; ++j) {
      v8f s = zero8();
      bfrag kf[4];
#pragma unroll
      for (int c = 0; c < 4; ++c)
        kf[c] = load_frag_b(&kt[j * 16][c * 32], KT_LD, lane);
#pragma unroll
      for (int c = 0; c < 4; ++c) s = wmma_bf16(qf[c], kf[c], s);
      sc[j] = s;
    }

    if (!linear_mode) {
      // scale + additive mask (C-layout: row = v+8*half, col = l15)
#pragma unroll
      for (int j = 0; j < 4; ++j) {
        const int ki = kb + j * 16 + l15;
#pragma unroll
        for (int v = 0; v < 8; ++v) {
          const int qi = q0 + v + 8 * half;
          const bool ok =
              (ki <= qi) && (!band || (qi - ki) < LAST_ || ki < FIRST_);
          sc[j][v] = ok ? sc[j][v] * SCALE_ : -1e30f;
        }
      }
      // online softmax update
#pragma unroll
      for (int v = 0; v < 8; ++v) {
        float rm = fmaxf(fmaxf(sc[0][v], sc[1][v]), fmaxf(sc[2][v], sc[3][v]));
#pragma unroll
        for (int m = 1; m < 16; m <<= 1) rm = fmaxf(rm, __shfl_xor(rm, m, 32));
        const float nm = fmaxf(mreg[v], rm);
        const float scale = __expf(mreg[v] - nm);
        mreg[v] = nm;
        float rs = 0.0f;
#pragma unroll
        for (int j = 0; j < 4; ++j) {
          const float p = __expf(sc[j][v] - nm);
          sc[j][v] = p;
          rs += p;
        }
#pragma unroll
        for (int m = 1; m < 16; m <<= 1) rs += __shfl_xor(rs, m, 32);
        lreg[v] = lreg[v] * scale + rs;
#pragma unroll
        for (int dt = 0; dt < 8; ++dt) acc[dt][v] = acc[dt][v] * scale;
      }
    } else {
      // linear branch: causal zero-mask, no softmax
#pragma unroll
      for (int j = 0; j < 4; ++j) {
        const int ki = kb + j * 16 + l15;
#pragma unroll
        for (int v = 0; v < 8; ++v) {
          const int qi = q0 + v + 8 * half;
          sc[j][v] = (ki <= qi) ? sc[j][v] * SCALE_ : 0.0f;
        }
      }
    }

    // P (C-layout) -> LDS row-major bf16 (per-wave region)
#pragma unroll
    for (int j = 0; j < 4; ++j)
#pragma unroll
      for (int v = 0; v < 8; ++v)
        pp[wave][v + 8 * half][j * 16 + l15] = f2bf(sc[j][v]);

    // PV: acc[dt] += P(16x64) x V(64x16), ki split in two 32-chunks,
    // B-fragments batched 4 at a time so one dscnt wait covers 4 WMMAs.
#pragma unroll
    for (int c2 = 0; c2 < 2; ++c2) {
      bfrag pa = load_frag_a(&pp[wave][0][c2 * 32], P_LD, lane);
#pragma unroll
      for (int g = 0; g < 2; ++g) {
        bfrag vb[4];
#pragma unroll
        for (int t = 0; t < 4; ++t)
          vb[t] = load_frag_b(&vt[(g * 4 + t) * 16][c2 * 32], VT_LD, lane);
#pragma unroll
        for (int t = 0; t < 4; ++t)
          acc[g * 4 + t] = wmma_bf16(pa, vb[t], acc[g * 4 + t]);
      }
    }
  }

  // epilogue
  if (!linear_mode) {
    const int slot = perm[h];   // inv_perm[slot] == h  =>  head h lands in slot perm[h]
#pragma unroll
    for (int v = 0; v < 8; ++v) {
      const float inv = 1.0f / lreg[v];
      const int q = q0 + v + 8 * half;
      const size_t base = (((size_t)b * S_ + q) * H_ + slot) * D_;
#pragma unroll
      for (int dt = 0; dt < 8; ++dt)
        ctx[base + dt * 16 + l15] = f2bf(acc[dt][v] * inv);
    }
  } else {
    const int th = h - ATTNH_;
#pragma unroll
    for (int v = 0; v < 8; ++v) {
      const int q = q0 + v + 8 * half;
      const size_t base = (((size_t)b * S_ + q) * TNH_ + th) * D_;
#pragma unroll
      for (int dt = 0; dt < 8; ++dt)
        out_tn[base + dt * 16 + l15] = acc[dt][v];
    }
  }
}

// ---------------- RMSNorm over the 1024-dim linear concat -> ctx (perm slots) ----------------
__global__ __launch_bounds__(256)
void tnorm_rms_kernel(const float* __restrict__ out_tn,
                      const int* __restrict__ perm,
                      unsigned short* __restrict__ ctx) {
  __shared__ float red[256];
  const int bs = blockIdx.x;             // b*S + s
  const int tid = threadIdx.x;
  const float* row = out_tn + (size_t)bs * (TNH_ * D_);
  float ss = 0.0f;
#pragma unroll
  for (int i = tid; i < TNH_ * D_; i += 256) { const float x = row[i]; ss += x * x; }
  red[tid] = ss;
  __syncthreads();
  for (int off = 128; off > 0; off >>= 1) {
    if (tid < off) red[tid] += red[tid + off];
    __syncthreads();
  }
  const float inv = rsqrtf(red[0] / (float)(TNH_ * D_) + 1e-6f);
  for (int i = tid; i < TNH_ * D_; i += 256) {
    const int th = i >> 7, d = i & 127;
    ctx[((size_t)bs * H_ + perm[ATTNH_ + th]) * D_ + d] = f2bf(row[i] * inv);
  }
}

// ---------------- host launcher ----------------
extern "C" void kernel_launch(void* const* d_in, const int* in_sizes, int n_in,
                              void* d_out, int out_size, void* d_ws, size_t ws_size,
                              hipStream_t stream) {
  (void)in_sizes; (void)n_in; (void)out_size;
  const float* hidden = (const float*)d_in[0];
  const float* q_w = (const float*)d_in[1];
  const float* q_b = (const float*)d_in[2];
  const float* k_w = (const float*)d_in[3];
  const float* k_b = (const float*)d_in[4];
  const float* v_w = (const float*)d_in[5];
  const float* v_b = (const float*)d_in[6];
  const float* o_w = (const float*)d_in[7];
  const unsigned char* policy = (const unsigned char*)d_in[8];
  const int* perm = (const int*)d_in[9];
  float* out = (float*)d_out;

  char* ws = (char*)d_ws;
  size_t off = 0;
  auto alloc = [&](size_t bytes) -> void* {
    void* p = ws + off;
    off += (bytes + 255) & ~(size_t)255;
    return p;
  };
  const size_t MS = (size_t)B_ * S_;                 // 4096 rows
  unsigned short* Xb  = (unsigned short*)alloc(MS * HID_ * 2);
  unsigned short* Wq  = (unsigned short*)alloc((size_t)HID_ * HID_ * 2);
  unsigned short* Wk  = (unsigned short*)alloc((size_t)KVH_ * D_ * HID_ * 2);
  unsigned short* Wv  = (unsigned short*)alloc((size_t)KVH_ * D_ * HID_ * 2);
  unsigned short* Wo  = (unsigned short*)alloc((size_t)HID_ * HID_ * 2);
  unsigned short* Qg  = (unsigned short*)alloc(MS * HID_ * 2);
  unsigned short* Kg  = (unsigned short*)alloc(MS * KVH_ * D_ * 2);
  unsigned short* Vg  = (unsigned short*)alloc(MS * KVH_ * D_ * 2);
  unsigned short* Qp  = (unsigned short*)alloc(MS * HID_ * 2);
  unsigned short* Kr  = (unsigned short*)alloc(MS * KVH_ * D_ * 2);
  unsigned short* ctx = (unsigned short*)alloc(MS * HID_ * 2);
  float* out_tn       = (float*)alloc(MS * TNH_ * D_ * 4);
  if (off > ws_size) return;  // insufficient scratch

  // 1) fp32 -> bf16 packs
  auto pk = [&](const float* s, unsigned short* dd, int n) {
    pack_bf16_kernel<<<(n + 255) / 256, 256, 0, stream>>>(s, dd, n);
  };
  pk(hidden, Xb, B_ * S_ * HID_);
  pk(q_w, Wq, HID_ * HID_);
  pk(k_w, Wk, KVH_ * D_ * HID_);
  pk(v_w, Wv, KVH_ * D_ * HID_);
  pk(o_w, Wo, HID_ * HID_);

  // 2) QKV projections (bf16 WMMA, bf16 out with bias)
  dim3 gq(HID_ / 64, (B_ * S_) / 128);
  gemm_wmma_kernel<true><<<gq, 256, 0, stream>>>(Xb, Wq, q_b, Qg, nullptr,
                                                 B_ * S_, HID_, HID_);
  dim3 gkv((KVH_ * D_) / 64, (B_ * S_) / 128);
  gemm_wmma_kernel<true><<<gkv, 256, 0, stream>>>(Xb, Wk, k_b, Kg, nullptr,
                                                  B_ * S_, KVH_ * D_, HID_);
  gemm_wmma_kernel<true><<<gkv, 256, 0, stream>>>(Xb, Wv, v_b, Vg, nullptr,
                                                  B_ * S_, KVH_ * D_, HID_);

  // 3) RoPE + head permutation
  rope_permute_kernel<<<dim3(S_, B_), 128, 0, stream>>>(Qg, Kg, perm, Qp, Kr);

  // 4) attention: softmax heads (mode 0) and linear heads (mode 1)
  attn_wmma_kernel<<<dim3(S_ / 128, ATTNH_, B_), 256, 0, stream>>>(
      Qp, Kr, Vg, policy, perm, ctx, out_tn, 0);
  attn_wmma_kernel<<<dim3(S_ / 128, TNH_, B_), 256, 0, stream>>>(
      Qp, Kr, Vg, policy, perm, ctx, out_tn, 1);

  // 5) RMSNorm of linear concat into ctx (inverse-permuted slots)
  tnorm_rms_kernel<<<B_ * S_, 256, 0, stream>>>(out_tn, perm, ctx);

  // 6) output projection -> fp32 d_out
  gemm_wmma_kernel<false><<<dim3(HID_ / 64, (B_ * S_) / 128), 256, 0, stream>>>(
      ctx, Wo, nullptr, nullptr, out, B_ * S_, HID_, HID_);
}
